// PixelCNN_10960756540235
// MI455X (gfx1250) — matrix operand order repaired and checked by reading, not compiled, not run on hardware
//
#include <hip/hip_runtime.h>
#include <hip/hip_bf16.h>

typedef __attribute__((ext_vector_type(16))) _Float16 v16h;
typedef __attribute__((ext_vector_type(8)))  float    v8f;
typedef unsigned int u32x4 __attribute__((ext_vector_type(4)));
typedef int          i32x4 __attribute__((ext_vector_type(4)));
typedef int          i32x8 __attribute__((ext_vector_type(8)));

#define NB  256
#define NH  64
#define NW  64
#define NC  64
#define ACT_HALVES ((size_t)NB * NH * NW * NC)        // 67,108,864 halves (128 MB fp16)
#define PACK_PER_LAYER 20480                          // 5 taps * 2 Kc * 4 Mt * 32 lanes * 16 halves

struct S32B { uint4 a, b; };                          // 32-byte chunk, 16B-aligned

// ---------------------------------------------------------------------------
// Pack W1..W4 (f32 [64][64][3][3], OIHW) into fp16 A-fragments laid out exactly
// per lane for v_wmma_f32_16x16x32_f16 in contiguous k order (same order as the B reads):
//   lane l: m = l&15, K-half = l>>4 ;  half h: K = (l>>4)*16 + h
// Only the 5 unmasked (type-B) taps are emitted -> mask applied implicitly.
// ---------------------------------------------------------------------------
__global__ void __launch_bounds__(256) pack_w_kernel(
    const float* __restrict__ W1, const float* __restrict__ W2,
    const float* __restrict__ W3, const float* __restrict__ W4,
    _Float16* __restrict__ packed)
{
    int i = blockIdx.x * 256 + threadIdx.x;
    if (i >= 4 * PACK_PER_LAYER) return;
    int L = i / PACK_PER_LAYER;
    int r = i % PACK_PER_LAYER;
    int h    =  r        & 15;
    int lane = (r >> 4)  & 31;
    int mt   = (r >> 9)  & 3;
    int kc   = (r >> 11) & 1;
    int t    =  r >> 12;                 // 0..4
    const float* W = (L == 0) ? W1 : (L == 1) ? W2 : (L == 2) ? W3 : W4;
    constexpr int TKY[5] = {0, 0, 0, 1, 1};
    constexpr int TKX[5] = {0, 1, 2, 0, 1};
    int oc = mt * 16 + (lane & 15);
    int K  = ((lane >> 4) * 16) + h;
    int ic = kc * 32 + K;
    packed[i] = (_Float16)W[(oc * 64 + ic) * 9 + TKY[t] * 3 + TKX[t]];
}

// ---------------------------------------------------------------------------
// Conv0: masked 7x7 type-A, 1 input channel, x = value*2-1, ReLU, -> NHWC fp16.
// 24 live taps (rows 0..2 full, row 3 cols 0..2). One thread per pixel.
// ---------------------------------------------------------------------------
__global__ void __launch_bounds__(256) conv0_kernel(
    const float* __restrict__ value, const float* __restrict__ W0,
    const float* __restrict__ b0, _Float16* __restrict__ out)
{
    __shared__ float wsm[24 * 64];
    const int tid = threadIdx.x;
    for (int i = tid; i < 24 * 64; i += 256) {
        int t = i >> 6, oc = i & 63;
        int ky = (t < 21) ? (t / 7) : 3;
        int kx = (t < 21) ? (t % 7) : (t - 21);
        wsm[t * 64 + oc] = W0[oc * 49 + ky * 7 + kx];
    }
    __syncthreads();

    const int p = blockIdx.x * 256 + tid;
    const int b = p >> 12, rem = p & 4095, y = rem >> 6, x = rem & 63;

    float xv[24];
#pragma unroll
    for (int t = 0; t < 24; ++t) {
        int ky = (t < 21) ? (t / 7) : 3;
        int kx = (t < 21) ? (t % 7) : (t - 21);
        int yy = y + ky - 3, xx = x + kx - 3;
        float v = 0.f;
        if (yy >= 0 && (unsigned)xx < 64u)            // yy <= y <= 63 always
            v = value[(b << 12) + (yy << 6) + xx] * 2.f - 1.f;
        xv[t] = v;
    }

    const size_t obase = (size_t)p * 64;
#pragma unroll
    for (int ocb = 0; ocb < 8; ++ocb) {
        float acc[8];
#pragma unroll
        for (int j = 0; j < 8; ++j) acc[j] = b0[ocb * 8 + j];
#pragma unroll
        for (int t = 0; t < 24; ++t) {
#pragma unroll
            for (int j = 0; j < 8; ++j)
                acc[j] = fmaf(wsm[t * 64 + ocb * 8 + j], xv[t], acc[j]);
        }
        union { uint4 u; _Float16 h[8]; } o;
#pragma unroll
        for (int j = 0; j < 8; ++j) o.h[j] = (_Float16)fmaxf(acc[j], 0.f);
        *(uint4*)(out + obase + ocb * 8) = o.u;
    }
}

// ---------------------------------------------------------------------------
// TDM: async-copy one image row (64 px * 64 ch fp16, contiguous in NHWC) into
// LDS with a 4-DWORD pad after every 32 DWORDs (-> 144 B pixel stride that
// matches the compute-side 72-half layout). D# per ISA ch.8:
//   group0: count=1 | lds_addr | global_addr[56:0] | type=2
//   group1: data_size=4B, pad_enable, pad_interval=32dw, pad_amount=4dw,
//           tensor_dim0=32dw, tensor_dim1=64, tile 32x64, stride0=32dw
// Toolchain uses the 6-arg builtin: (g0, g1, g2, g3, g4, cpol); groups 2..4
// are unused for a <=2D tensor -> zero-filled.
// ---------------------------------------------------------------------------
__device__ __forceinline__ void tdm_load_row(unsigned int lds_byte_addr,
                                             const _Float16* gsrc)
{
    unsigned long long ga = (unsigned long long)(uintptr_t)gsrc;
    u32x4 g0;
    g0[0] = 1u;                                                  // count=1
    g0[1] = lds_byte_addr;                                       // lds_addr
    g0[2] = (unsigned int)ga;                                    // global_addr lo
    g0[3] = (unsigned int)((ga >> 32) & 0x01FFFFFFu) | (2u << 30); // ga hi | type=2
    i32x8 g1;
    g1[0] = (int)((2u << 16)        // data_size = 4 B
                | (1u << 20)        // pad_enable
                | (4u << 22)        // pad_interval: 2^(4+1)=32 DWORDs
                | (3u << 25));      // pad_amount: 3+1 = 4 DWORDs
    g1[1] = (int)(32u << 16);       // tensor_dim0[15:0] = 32 DWORDs / pixel
    g1[2] = (int)(64u << 16);       // tensor_dim1[15:0] = 64 pixels
    g1[3] = (int)(32u << 16);       // tile_dim0 = 32 DWORDs
    g1[4] = 64;                     // tile_dim1 = 64 pixels (tile_dim2 = 0)
    g1[5] = 32;                     // tensor_dim0_stride = 32 DWORDs (contig)
    g1[6] = 0;
    g1[7] = 0;
    i32x4 z4 = {0, 0, 0, 0};        // groups 2/3 unused (<=2D tensor)
    i32x8 z8 = {0, 0, 0, 0, 0, 0, 0, 0};
    __builtin_amdgcn_tensor_load_to_lds(g0, g1, z4, z4, z8, 0);
}

// ---------------------------------------------------------------------------
// Masked 3x3 (type B) conv as implicit GEMM with WMMA f16->f32.
// One workgroup = one image row (64 px, all 64 oc). 8 waves: wave w computes
// oc-tile (w>>1) for pixel tiles {2*(w&1), 2*(w&1)+1}. K = 5 taps x 64 ic,
// consumed 32 at a time -> 10 WMMAs per accumulator, 20 per wave.
// Rows y-1 and y staged in LDS by the Tensor Data Mover (x-halos are always
// outside the image -> zero-filled with ds_stores). Pixel stride 72 halves.
// ---------------------------------------------------------------------------
__global__ void __launch_bounds__(256) conv3x3_wmma_kernel(
    const _Float16* __restrict__ in, const _Float16* __restrict__ packedA,
    const float* __restrict__ bias, _Float16* __restrict__ out)
{
    __shared__ uint4 lds4[2 * 66 * 9];                 // 19,008 B
    _Float16* smem = (_Float16*)lds4;

    const int tid = threadIdx.x;
    const int bid = blockIdx.x;                        // NB*NH = 16384
    const int b = bid >> 6, y = bid & 63;
    const unsigned int lds_base = (unsigned int)(uintptr_t)(&lds4[0]);

    // ---- zero the always-OOB x-halo pixels (px 0 and 65, both rows) ----
    if (tid < 32) {
        int r = tid >> 4, px = ((tid >> 3) & 1) ? 65 : 0, sub = tid & 7;
        uint4 z = {0u, 0u, 0u, 0u};
        lds4[(r * 66 + px) * 9 + sub] = z;
    }
    // ---- top edge: whole row y-1 is zero when y == 0 ----
    if (y == 0) {
        uint4 z = {0u, 0u, 0u, 0u};
        for (int i = tid; i < 594; i += 256) lds4[i] = z;
    }
    // ---- wave 0 issues the async TDM row copies (EXEC ignored; 1 issue/wave)
    if (tid < 32) {
        if (y > 0)
            tdm_load_row(lds_base + (0 * 594 + 9) * 16,
                         in + (size_t)((b * 64 + (y - 1)) * 64) * 64);
        tdm_load_row(lds_base + (1 * 594 + 9) * 16,
                     in + (size_t)((b * 64 + y) * 64) * 64);
        __builtin_amdgcn_s_wait_tensorcnt(0);
    }
    __syncthreads();

    const int lane = tid & 31, wid = tid >> 5;
    const int mt  = wid >> 1;
    const int nt0 = (wid & 1) * 2;
    const int lm = lane & 15, lh = lane >> 4;

    v8f acc0 = {}, acc1 = {};

    constexpr int TROW[5] = {0, 0, 0, 1, 1};           // dy=-1 -> row0, dy=0 -> row1
    constexpr int TDX[5]  = {-1, 0, 1, -1, 0};
#pragma unroll
    for (int t = 0; t < 5; ++t) {
#pragma unroll
        for (int kc = 0; kc < 2; ++kc) {
            // A fragment: prepacked, lane-contiguous 32B, L2-resident
            const int fidx = (((t * 2 + kc) * 4 + mt) * 32 + lane) * 16;
            v16h a = *(const v16h*)(packedA + fidx);

            const int chb = kc * 32 + lh * 16;         // 16 consecutive channels
            const int px0 = nt0 * 16 + lm + TDX[t] + 1;
            const int off0 = TROW[t] * 4752 + px0 * 72 + chb;

            S32B s0 = *(const S32B*)(smem + off0);
            v16h bf0 = __builtin_bit_cast(v16h, s0);
            acc0 = __builtin_amdgcn_wmma_f32_16x16x32_f16(
                false, a, false, bf0, (short)0, acc0, false, false);

            S32B s1 = *(const S32B*)(smem + off0 + 16 * 72);
            v16h bf1 = __builtin_bit_cast(v16h, s1);
            acc1 = __builtin_amdgcn_wmma_f32_16x16x32_f16(
                false, a, false, bf1, (short)0, acc1, false, false);
        }
    }

    // ---- epilogue: bias + ReLU, fp16 NHWC store (8 contiguous oc per lane) ----
    const int ocb = mt * 16 + lh * 8;                  // D layout: m = r + 8*(l>>4)
    float bs[8];
#pragma unroll
    for (int r = 0; r < 8; ++r) bs[r] = bias[ocb + r];

    union { uint4 u; _Float16 h[8]; } o0, o1;
#pragma unroll
    for (int r = 0; r < 8; ++r) {
        o0.h[r] = (_Float16)fmaxf(acc0[r] + bs[r], 0.f);
        o1.h[r] = (_Float16)fmaxf(acc1[r] + bs[r], 0.f);
    }
    const size_t rowbase = (size_t)((b * 64 + y) * 64) * 64;
    *(uint4*)(out + rowbase + (size_t)(nt0 * 16 + lm) * 64 + ocb)       = o0.u;
    *(uint4*)(out + rowbase + (size_t)((nt0 + 1) * 16 + lm) * 64 + ocb) = o1.u;
}

// ---------------------------------------------------------------------------
// Head: z = <h, Wo> + bo ; lp = v*z - softplus(z) ; sum over pixels per image.
// ---------------------------------------------------------------------------
__global__ void __launch_bounds__(256) head_kernel(
    const _Float16* __restrict__ hin, const float* __restrict__ value,
    const float* __restrict__ Wo, const float* __restrict__ bo,
    float* __restrict__ outp)
{
    __shared__ float wsm[64];
    __shared__ float red[256];
    const int tid = threadIdx.x, b = blockIdx.x;
    if (tid < 64) wsm[tid] = Wo[tid];
    __syncthreads();

    float lp = 0.f;
    const float bias = bo[0];
    for (int i = tid; i < 4096; i += 256) {
        const uint4* hp = (const uint4*)(hin + ((size_t)b * 4096 + i) * 64);
        float z = bias;
#pragma unroll
        for (int c8 = 0; c8 < 8; ++c8) {
            union { uint4 u; _Float16 h[8]; } d;
            d.u = hp[c8];
#pragma unroll
            for (int j = 0; j < 8; ++j)
                z = fmaf((float)d.h[j], wsm[c8 * 8 + j], z);
        }
        float v  = value[b * 4096 + i];
        float sp = fmaxf(z, 0.f) + log1pf(expf(-fabsf(z)));
        lp += v * z - sp;
    }
    red[tid] = lp;
    __syncthreads();
    for (int s = 128; s > 0; s >>= 1) {
        if (tid < s) red[tid] += red[tid + s];
        __syncthreads();
    }
    if (tid == 0) outp[b] = red[0];
}

// ---------------------------------------------------------------------------
extern "C" void kernel_launch(void* const* d_in, const int* in_sizes, int n_in,
                              void* d_out, int out_size, void* d_ws, size_t ws_size,
                              hipStream_t stream)
{
    const float* value = (const float*)d_in[0];
    const float* W0    = (const float*)d_in[1];
    const float* b0    = (const float*)d_in[2];
    const float* Wc[4] = {(const float*)d_in[3], (const float*)d_in[5],
                          (const float*)d_in[7], (const float*)d_in[9]};
    const float* bc[4] = {(const float*)d_in[4], (const float*)d_in[6],
                          (const float*)d_in[8], (const float*)d_in[10]};
    const float* Wo    = (const float*)d_in[11];
    const float* bo    = (const float*)d_in[12];
    float* out = (float*)d_out;

    _Float16* hA     = (_Float16*)d_ws;                // ping (128 MB)
    _Float16* hB     = hA + ACT_HALVES;                // pong (128 MB)
    _Float16* packed = hB + ACT_HALVES;                // 4 * 40 KB A-fragments

    pack_w_kernel<<<(4 * PACK_PER_LAYER + 255) / 256, 256, 0, stream>>>(
        Wc[0], Wc[1], Wc[2], Wc[3], packed);

    conv0_kernel<<<(NB * NH * NW) / 256, 256, 0, stream>>>(value, W0, b0, hA);

    _Float16* cin = hA; _Float16* cout = hB;
    for (int L = 0; L < 4; ++L) {
        conv3x3_wmma_kernel<<<NB * NH, 256, 0, stream>>>(
            cin, packed + (size_t)L * PACK_PER_LAYER, bc[L], cout);
        _Float16* t = cin; cin = cout; cout = t;
    }

    head_kernel<<<NB, 256, 0, stream>>>(cin, value, Wo, bo, out);
}
